// KeyValueMemoryNetwork_62113817035231
// MI455X (gfx1250) — matrix operand order repaired and checked by reading, not compiled, not run on hardware
//
#include <hip/hip_runtime.h>
#include <hip/hip_bf16.h>

typedef __attribute__((ext_vector_type(16))) _Float16 v16h;
typedef __attribute__((ext_vector_type(8)))  float    v8f;

constexpr int B_ = 8;
constexpr int H_ = 256;
constexpr int K_ = 256;
constexpr int E_ = 128;
constexpr int FVOCAB_ = 1000;
constexpr int HT = 16;               // h rows handled per block

// Phase 1: per (b, 16-row h tile):
//   u = hidden * gathered-keys^T via v_wmma_f32_16x16x32_f16 (f32 accum)
//   delta = exp(u/sqrt(E)) * mask  -> LDS
//   p = delta / (rowsum + 1e-10)   -> LDS (ds_add_f32 rowsum)
//   o[h,:] = sum_k p[h,k] * value_emb[value_seq[b,h,k],:] -> workspace
__global__ __launch_bounds__(256)
void kvmn_attn_kernel(const float* __restrict__ hidden,
                      const float* __restrict__ key_emb,
                      const float* __restrict__ value_emb,
                      const int*   __restrict__ key_seq,
                      const int*   __restrict__ value_seq,
                      const int*   __restrict__ mask,
                      float* __restrict__ o_ws)
{
    __shared__ float p_lds[HT * K_];   // 16 KB: delta, then p (in place)
    __shared__ float rowsum[HT];

    const int blk = blockIdx.x;
    const int b   = blk >> 4;          // 16 h-tiles per batch element
    const int h0  = (blk & 15) * HT;

    const int tid  = threadIdx.x;
    const int lane = tid & 31;
    const int wv   = tid >> 5;         // wave 0..7
    const int half = lane >> 4;        // lane group within wave
    const int l16  = lane & 15;

    // Pull the (L2-resident) 500KB value table toward the caches early:
    // one 64B line per thread across the grid -> global_prefetch_b8.
    {
        const long line = (long)blk * 256 + tid;
        if (line * 64 < (long)FVOCAB_ * E_ * 4)
            __builtin_prefetch(value_emb + line * 16, 0, 3);
    }

    // ---- A fragments: hidden[b, h0 + (lane&15), :] as f16, 4 chunks of 32 ----
    // ISA 16-bit A 16x32 layout: lane m holds row m; half 0 -> K 0..7 & 16..23,
    // half 1 -> K 8..15 & 24..31 (elements 0..7 then 8..15).
    v16h afrag[4];
    {
        const float* hrow = hidden + ((size_t)b * H_ + h0 + l16) * E_;
        for (int ch = 0; ch < 4; ++ch) {
            const int eb = ch * 32 + half * 8;
            for (int j = 0; j < 8; ++j) {
                afrag[ch][j]     = (_Float16)hrow[eb + j];
                afrag[ch][8 + j] = (_Float16)hrow[eb + 16 + j];
            }
        }
    }

    const float inv_scale = 0.08838834764831845f;  // 1/sqrt(128)

    // ---- each wave computes two 16x16 u-tiles (K/16 = 16 tiles / 8 waves) ----
    for (int tt = 0; tt < 2; ++tt) {
        const int t = wv * 2 + tt;
        const int k = t * 16 + l16;                       // this lane's k column
        const int ki = key_seq[b * K_ + k];
        const float* krow = key_emb + (size_t)ki * E_;

        v8f c = {};
        for (int ch = 0; ch < 4; ++ch) {
            // ISA 16-bit B 32x16 layout: lane n holds col n; half 0 -> K 0..7 & 16..23,
            // half 1 -> K 8..15 & 24..31 (same order as A).
            const int eb = ch * 32 + half * 8;
            v16h bfrag;
            for (int j = 0; j < 8; ++j) {
                bfrag[j]     = (_Float16)krow[eb + j];
                bfrag[8 + j] = (_Float16)krow[eb + 16 + j];
            }
            c = __builtin_amdgcn_wmma_f32_16x16x32_f16(
                    false, afrag[ch], false, bfrag, (short)0, c, false, false);
        }

        // C layout: element r -> row M = r + 8*half, col N = lane&15.
        for (int r = 0; r < 8; ++r) {
            const int m = r + half * 8;
            const float u  = c[r] * inv_scale;
            const int  mk  = mask[((size_t)b * H_ + h0 + m) * K_ + k];
            p_lds[m * K_ + k] = mk ? __expf(u) : 0.0f;
        }
    }

    if (tid < HT) rowsum[tid] = 0.0f;
    __syncthreads();

    // ---- row sums: 16 threads per row, ds_add_f32 reduction ----
    {
        const int r  = tid >> 4;
        const int jj = tid & 15;
        float s = 0.0f;
        for (int q = 0; q < 16; ++q) s += p_lds[r * K_ + jj + q * 16];
        atomicAdd(&rowsum[r], s);
    }
    __syncthreads();

    // ---- normalize in place: p = delta / (sum + 1e-10) ----
    {
        const int r  = tid >> 4;
        const int jj = tid & 15;
        const float inv = 1.0f / (rowsum[r] + 1e-10f);
        for (int q = 0; q < 16; ++q) p_lds[r * K_ + jj + q * 16] *= inv;
    }
    __syncthreads();

    // ---- value aggregation: wave w owns rows 2w, 2w+1; lane owns 4 columns ----
    for (int rr = 0; rr < 2; ++rr) {
        const int hl = wv * 2 + rr;
        const int h  = h0 + hl;
        const int* vs = value_seq + ((size_t)b * H_ + h) * K_;
        float4 acc = make_float4(0.f, 0.f, 0.f, 0.f);
        for (int k = 0; k < K_; ++k) {
            const float pk = p_lds[hl * K_ + k];   // LDS broadcast, wave-uniform
            if (pk != 0.0f) {                      // uniform branch: skip masked k
                const float4 vv = *(const float4*)(value_emb +
                                      (size_t)vs[k] * E_ + lane * 4);
                acc.x += pk * vv.x; acc.y += pk * vv.y;
                acc.z += pk * vv.z; acc.w += pk * vv.w;
            }
        }
        *(float4*)(o_ws + ((size_t)b * H_ + h) * E_ + lane * 4) = acc;
    }
}

// Phase 2: avg over H with nonzero count (matches reference semantics exactly).
__global__ __launch_bounds__(256)
void kvmn_reduce_kernel(const float* __restrict__ o_ws, float* __restrict__ out)
{
    const int i = blockIdx.x * blockDim.x + threadIdx.x;   // 0 .. B*E-1
    if (i >= B_ * E_) return;
    const int b = i / E_;
    const int e = i % E_;
    float s = 0.0f, cnt = 0.0f;
    for (int h = 0; h < H_; ++h) {
        const float v = o_ws[((size_t)b * H_ + h) * E_ + e];
        s   += v;
        cnt += (v != 0.0f) ? 1.0f : 0.0f;
    }
    out[i] = s / cnt;
}

extern "C" void kernel_launch(void* const* d_in, const int* in_sizes, int n_in,
                              void* d_out, int out_size, void* d_ws, size_t ws_size,
                              hipStream_t stream) {
    const float* hidden    = (const float*)d_in[0];
    const float* key_emb   = (const float*)d_in[1];
    const float* value_emb = (const float*)d_in[2];
    const int*   key_seq   = (const int*)d_in[3];
    const int*   value_seq = (const int*)d_in[4];
    const int*   mask      = (const int*)d_in[5];
    float* out  = (float*)d_out;
    float* o_ws = (float*)d_ws;   // B*H*E floats = 1 MB scratch

    kvmn_attn_kernel<<<B_ * (H_ / HT), 256, 0, stream>>>(
        hidden, key_emb, value_emb, key_seq, value_seq, mask, o_ws);
    kvmn_reduce_kernel<<<(B_ * E_ + 255) / 256, 256, 0, stream>>>(o_ws, out);
}
